// TransformerSentenceEncoder_84576495993362
// MI455X (gfx1250) — compile-verified, matrix operand order checked by reading, not run on hardware
//
#include <hip/hip_runtime.h>
#include <hip/hip_bf16.h>
#include <cstdint>
#include <cstddef>

// ---------------------------------------------------------------------------
// TUPE transformer encoder on gfx1250 (wave32, WMMA 16x16x32 f16, f32 accum)
// ---------------------------------------------------------------------------

typedef __attribute__((ext_vector_type(16))) _Float16 v16h;
typedef __attribute__((ext_vector_type(8)))  float    v8f;
typedef int gv4i __attribute__((vector_size(16)));   // matches builtin's v4i

#define Dm   768
#define Sm   256
#define Hm   8
#define HDm  96
#define Bm   32
#define Lm   6
#define NTOK (Bm * Sm)
#define ATT_SCALE 0.07216878364870323f   // (HD*2)^-0.5 = 192^-0.5

#if __has_builtin(__builtin_amdgcn_global_load_async_to_lds_b128) && \
    __has_builtin(__builtin_amdgcn_s_wait_asynccnt)
#define USE_ASYNC_LDS 1
#else
#define USE_ASYNC_LDS 0
#endif

// Copy 16B global -> LDS. Async (ASYNCcnt-tracked, no staging VGPRs) when the
// gfx1250 async builtins exist; plain load+ds_store otherwise.
__device__ __forceinline__ void ld_tile16(const _Float16* g, _Float16* l) {
#if USE_ASYNC_LDS
  __builtin_amdgcn_global_load_async_to_lds_b128(
      (__attribute__((address_space(1))) gv4i*)(void*)g,
      (__attribute__((address_space(3))) gv4i*)l, 0, 0);
#else
  *(float4*)l = *(const float4*)g;
#endif
}

__device__ __forceinline__ void wait_async_4() {
#if USE_ASYNC_LDS
  __builtin_amdgcn_s_wait_asynccnt(4);   // previous 4-op batch complete
#endif
}
__device__ __forceinline__ void wait_async_0() {
#if USE_ASYNC_LDS
  __builtin_amdgcn_s_wait_asynccnt(0);   // all my async LDS writes complete
#endif
}

__device__ __forceinline__ void async_fence_workgroup() {
  wait_async_0();
  __syncthreads();
}

// Pack a 16x32 f16 A-fragment (or the mirrored B-fragment) for one lane.
// Lane l: row = l&15, off = (l>=16)?8:0; two 16B chunks at +off and +16+off.
__device__ __forceinline__ v16h pack16(const _Float16* p, int off) {
  v16h r;
  ((float4*)&r)[0] = *(const float4*)(p + off);
  ((float4*)&r)[1] = *(const float4*)(p + 16 + off);
  return r;
}

__device__ __forceinline__ v8f wmma_f16(v16h a, v16h b, v8f c) {
  return __builtin_amdgcn_wmma_f32_16x16x32_f16(false, a, false, b, (short)0, c,
                                                false, false);
}

// ---------------------------------------------------------------------------
// C[M,768] = (A[M,768] * W[768,768]^T + bias[768]) * alpha
// 128x128 tile per 256-thread (8-wave) workgroup; K stepped by 32 via LDS
// with ping-pong double buffering: async DMA for tile k+1 overlaps WMMA on
// tile k (s_wait_asynccnt 4 keeps the new batch in flight).
// F32OUT=false -> f16 output Ch ; F32OUT=true -> f32 output Cf.
// ---------------------------------------------------------------------------
template <bool F32OUT>
__global__ __launch_bounds__(256) void gemm_xwt(
    const _Float16* __restrict__ A, const _Float16* __restrict__ W,
    const float* __restrict__ bias, float alpha,
    float* __restrict__ Cf, _Float16* __restrict__ Ch)
{
  __shared__ _Float16 As[2][128][40];  // 32 K-halves + 16B pad, x2 buffers
  __shared__ _Float16 Ws[2][128][40];
  const int tid  = threadIdx.x;
  const int wave = tid >> 5, lane = tid & 31;
  const int m0 = blockIdx.y * 128, n0 = blockIdx.x * 128;
  const int mw = (wave & 3) * 32;   // 2 M-tiles per wave
  const int nw = (wave >> 2) * 64;  // 4 N-tiles per wave
  const int off = (lane >= 16) ? 8 : 0;
  const int r   = lane & 15;
  const int hl  = lane >> 4;
  v8f acc[2][4] = {};

  const int lrow = tid >> 1;
  const int lcol = (tid & 1) * 16;
  const _Float16* ga = A + (size_t)(m0 + lrow) * Dm + lcol;
  const _Float16* gw = W + (size_t)(n0 + lrow) * Dm + lcol;

  auto load_tiles = [&](int k0, int buf) {
    ld_tile16(ga + k0,     &As[buf][lrow][lcol]);
    ld_tile16(ga + k0 + 8, &As[buf][lrow][lcol + 8]);
    ld_tile16(gw + k0,     &Ws[buf][lrow][lcol]);
    ld_tile16(gw + k0 + 8, &Ws[buf][lrow][lcol + 8]);
    __builtin_prefetch(ga + k0 + 64, 0, 3);   // two tiles ahead -> L2
    __builtin_prefetch(gw + k0 + 64, 0, 3);
  };
  auto compute = [&](int buf) {
    v16h af[2], wf[4];
#pragma unroll
    for (int i = 0; i < 2; ++i) af[i] = pack16(&As[buf][mw + i * 16 + r][0], off);
#pragma unroll
    for (int j = 0; j < 4; ++j) wf[j] = pack16(&Ws[buf][nw + j * 16 + r][0], off);
#pragma unroll
    for (int i = 0; i < 2; ++i)
#pragma unroll
      for (int j = 0; j < 4; ++j)
        acc[i][j] = wmma_f16(af[i], wf[j], acc[i][j]);
  };

  load_tiles(0, 0);                       // prologue: fill buffer 0
  for (int k0 = 0; k0 < Dm; k0 += 64) {
    // sub-step A: load k0+32 -> buf1, compute buf0
    __syncthreads();                      // buf1 fully consumed (prev sub-B)
    load_tiles(k0 + 32, 1);
    wait_async_4();                       // buf0's batch done; buf1 in flight
    __syncthreads();
    compute(0);
    // sub-step B: load k0+64 -> buf0 (if any), compute buf1
    if (k0 + 64 < Dm) {
      __syncthreads();                    // buf0 fully consumed (sub-A)
      load_tiles(k0 + 64, 0);
      wait_async_4();                     // buf1's batch done; buf0 in flight
      __syncthreads();
      compute(1);
    } else {
      wait_async_0();
      __syncthreads();
      compute(1);
    }
  }

  // Epilogue: single base pointer + compile-time offsets (no spills).
  const size_t base = (size_t)(m0 + mw + 8 * hl) * Dm + (size_t)(n0 + nw + r);
  float bv[4];
#pragma unroll
  for (int j = 0; j < 4; ++j) bv[j] = bias[n0 + nw + j * 16 + r];
  if (F32OUT) {
    float* op = Cf + base;
#pragma unroll
    for (int i = 0; i < 2; ++i)
#pragma unroll
      for (int j = 0; j < 4; ++j)
#pragma unroll
        for (int rr = 0; rr < 8; ++rr)
          op[(size_t)(i * 16 + rr) * Dm + j * 16] =
              (acc[i][j][rr] + bv[j]) * alpha;
  } else {
    _Float16* op = Ch + base;
#pragma unroll
    for (int i = 0; i < 2; ++i)
#pragma unroll
      for (int j = 0; j < 4; ++j)
#pragma unroll
        for (int rr = 0; rr < 8; ++rr)
          op[(size_t)(i * 16 + rr) * Dm + j * 16] =
              (_Float16)((acc[i][j][rr] + bv[j]) * alpha);
  }
}

// ---------------------------------------------------------------------------
// bias[h,i,j] = sum_d pq[i,h,d]*pk[j,h,d] + rel_table[rp[i,j], h]
// One wave per 16x16 tile, K=96 => 3 WMMAs.
// ---------------------------------------------------------------------------
__global__ __launch_bounds__(32) void bias_abs_rel(
    const _Float16* __restrict__ pq, const _Float16* __restrict__ pk,
    const float* __restrict__ table, const int* __restrict__ rp,
    float* __restrict__ biasOut)
{
  const int lane = threadIdx.x;
  const int h  = blockIdx.z;
  const int i0 = blockIdx.y * 16, j0 = blockIdx.x * 16;
  const int off = (lane >= 16) ? 8 : 0;
  const int nn = lane & 15;
  const int hl = lane >> 4;
  v8f acc = {};
  const _Float16* ap = pq + (size_t)(i0 + nn) * Dm + h * HDm;
  const _Float16* bp = pk + (size_t)(j0 + nn) * Dm + h * HDm;
#pragma unroll
  for (int kk = 0; kk < 3; ++kk)
    acc = wmma_f16(pack16(ap + kk * 32, off), pack16(bp + kk * 32, off), acc);
#pragma unroll
  for (int rr = 0; rr < 8; ++rr) {
    const int i = i0 + rr + 8 * hl, j = j0 + nn;
    const float rel = table[rp[i * Sm + j] * Hm + h];
    biasOut[((size_t)h * Sm + i) * Sm + j] = acc[rr] + rel;
  }
}

// ---------------------------------------------------------------------------
// Fused attention: per (b, h, 64 q-rows) workgroup, 4 waves (16 q-rows each).
// Streams K/V in 64-row chunks through LDS; online softmax; P@V via WMMA.
// ---------------------------------------------------------------------------
__global__ __launch_bounds__(128) void attn_fused(
    const _Float16* __restrict__ Q, const _Float16* __restrict__ Kt,
    const _Float16* __restrict__ V, const float* __restrict__ bias,
    _Float16* __restrict__ O)
{
  __shared__ _Float16 Ks[64][104];      // K chunk, row-major [j][d]
  __shared__ _Float16 Vt[96][72];       // V chunk, transposed [d][j]
  __shared__ _Float16 Ps[4][16][72];    // per-wave P tile [m][j]
  const int tid  = threadIdx.x;
  const int wave = tid >> 5, lane = tid & 31;
  const int h = blockIdx.y, b = blockIdx.z;
  const int qb = blockIdx.x * 64 + wave * 16;
  const int rowBase = b * Sm;
  const int off = (lane >= 16) ? 8 : 0;
  const int nn = lane & 15;
  const int hl = lane >> 4;

  v16h qf[3];
  {
    const _Float16* qp = Q + (size_t)(rowBase + qb + nn) * Dm + h * HDm;
    qf[0] = pack16(qp, off);
    qf[1] = pack16(qp + 32, off);
    qf[2] = pack16(qp + 64, off);
  }

  float mrow[8], lrowv[8];
  v8f acc_o[6] = {};
#pragma unroll
  for (int rr = 0; rr < 8; ++rr) { mrow[rr] = -1e30f; lrowv[rr] = 0.0f; }

  for (int c = 0; c < 4; ++c) {
    const int j0 = c * 64;
    __syncthreads();
    {   // load K chunk (64 rows x 96 halves), async to LDS
      const int row = tid >> 1, col = (tid & 1) * 48;
      const _Float16* src =
          Kt + (size_t)(rowBase + j0 + row) * Dm + h * HDm + col;
#pragma unroll
      for (int e = 0; e < 6; ++e) ld_tile16(src + 8 * e, &Ks[row][col + 8 * e]);
    }
    {   // load V chunk transposed (scatter, stays synchronous)
      const int j = tid & 63, d0 = (tid >> 6) * 48;
      const _Float16* src = V + (size_t)(rowBase + j0 + j) * Dm + h * HDm + d0;
      _Float16 tmp[48];
#pragma unroll
      for (int e = 0; e < 6; ++e) ((float4*)tmp)[e] = ((const float4*)src)[e];
#pragma unroll
      for (int e = 0; e < 48; ++e) Vt[d0 + e][j] = tmp[e];
    }
    async_fence_workgroup();

    // scores for this chunk: 4 j-tiles, K = 96
    float sv[4][8];
#pragma unroll
    for (int jt = 0; jt < 4; ++jt) {
      v8f sacc = {};
#pragma unroll
      for (int kk = 0; kk < 3; ++kk) {
        v16h bf = pack16(&Ks[jt * 16 + nn][kk * 32], off);
        sacc = wmma_f16(qf[kk], bf, sacc);
      }
#pragma unroll
      for (int rr = 0; rr < 8; ++rr) {
        const int qi = qb + rr + 8 * hl;
        const int j  = j0 + jt * 16 + nn;
        sv[jt][rr] = sacc[rr] + bias[((size_t)h * Sm + qi) * Sm + j];
      }
    }

    // online softmax (row reductions across each 16-lane half)
#pragma unroll
    for (int rr = 0; rr < 8; ++rr) {
      float v = fmaxf(fmaxf(sv[0][rr], sv[1][rr]), fmaxf(sv[2][rr], sv[3][rr]));
      for (int m = 1; m < 16; m <<= 1) v = fmaxf(v, __shfl_xor(v, m, 32));
      const float mn   = fmaxf(mrow[rr], v);
      const float corr = __expf(mrow[rr] - mn);
      lrowv[rr] *= corr;
#pragma unroll
      for (int nt = 0; nt < 6; ++nt) acc_o[nt][rr] *= corr;
      float ps = 0.0f;
#pragma unroll
      for (int jt = 0; jt < 4; ++jt) {
        const float pv = __expf(sv[jt][rr] - mn);
        ps += pv;
        Ps[wave][rr + 8 * hl][jt * 16 + nn] = (_Float16)pv;
      }
      for (int m = 1; m < 16; m <<= 1) ps += __shfl_xor(ps, m, 32);
      lrowv[rr] += ps;
      mrow[rr] = mn;
    }
    asm volatile("s_wait_dscnt 0x0" ::: "memory");  // P writes -> P reads

    // P(16x64) @ Vchunk(64x96)
    v16h pf0 = pack16(&Ps[wave][nn][0], off);
    v16h pf1 = pack16(&Ps[wave][nn][32], off);
#pragma unroll
    for (int nt = 0; nt < 6; ++nt) {
      v16h vf0 = pack16(&Vt[nt * 16 + nn][0], off);
      v16h vf1 = pack16(&Vt[nt * 16 + nn][32], off);
      acc_o[nt] = wmma_f16(pf0, vf0, acc_o[nt]);
      acc_o[nt] = wmma_f16(pf1, vf1, acc_o[nt]);
    }
  }

#pragma unroll
  for (int rr = 0; rr < 8; ++rr) {
    const float inv = 1.0f / lrowv[rr];
    const int qi = qb + rr + 8 * hl;
#pragma unroll
    for (int nt = 0; nt < 6; ++nt)
      O[(size_t)(rowBase + qi) * Dm + h * HDm + nt * 16 + nn] =
          (_Float16)(acc_o[nt][rr] * inv);
  }
}

// ---------------------------------------------------------------------------
// Small helper kernels
// ---------------------------------------------------------------------------
__global__ __launch_bounds__(256) void posln_kernel(
    const float* __restrict__ pe, const float* __restrict__ g,
    const float* __restrict__ bb, _Float16* __restrict__ out)
{
  const int row = blockIdx.x;
  const float* x = pe + (size_t)row * Dm;
  float v[3], s = 0.0f, s2 = 0.0f;
#pragma unroll
  for (int i = 0; i < 3; ++i) {
    v[i] = x[threadIdx.x + i * 256];
    s += v[i]; s2 += v[i] * v[i];
  }
  for (int m = 1; m < 32; m <<= 1) {
    s  += __shfl_xor(s,  m, 32);
    s2 += __shfl_xor(s2, m, 32);
  }
  __shared__ float ps[8], ps2[8];
  const int wave = threadIdx.x >> 5, lane = threadIdx.x & 31;
  if (lane == 0) { ps[wave] = s; ps2[wave] = s2; }
  __syncthreads();
  s = 0.0f; s2 = 0.0f;
#pragma unroll
  for (int i = 0; i < 8; ++i) { s += ps[i]; s2 += ps2[i]; }
  const float mean = s / (float)Dm;
  const float var  = s2 / (float)Dm - mean * mean;
  const float rstd = rsqrtf(var + 1e-5f);
#pragma unroll
  for (int i = 0; i < 3; ++i) {
    const int cidx = threadIdx.x + i * 256;
    out[(size_t)row * Dm + cidx] =
        (_Float16)((v[i] - mean) * rstd * g[cidx] + bb[cidx]);
  }
}

__global__ __launch_bounds__(256) void cvt_f32_f16(
    const float* __restrict__ in, _Float16* __restrict__ out, int n)
{
  const int i = blockIdx.x * 256 + threadIdx.x;
  if (i < n) out[i] = (_Float16)in[i];
}

__global__ __launch_bounds__(256) void gather_embed(
    const int* __restrict__ tok, const float* __restrict__ emb,
    _Float16* __restrict__ x)
{
  const int row = blockIdx.x;
  const int t = tok[row];
  const float* src = emb + (size_t)t * Dm;
  for (int c = threadIdx.x; c < Dm; c += 256)
    x[(size_t)row * Dm + c] = (_Float16)src[c];
}

__global__ __launch_bounds__(256) void copy_row0(
    const float* __restrict__ xf, float* __restrict__ out)
{
  const int b = blockIdx.x;
  for (int c = threadIdx.x; c < Dm; c += 256)
    out[(size_t)b * Dm + c] = xf[(size_t)(b * Sm) * Dm + c];
}

// ---------------------------------------------------------------------------
// Host orchestration
// ---------------------------------------------------------------------------
extern "C" void kernel_launch(void* const* d_in, const int* in_sizes, int n_in,
                              void* d_out, int out_size, void* d_ws, size_t ws_size,
                              hipStream_t stream)
{
  (void)in_sizes; (void)n_in; (void)out_size; (void)ws_size;

  const int*   tokens  = (const int*)  d_in[0];
  const float* emb     = (const float*)d_in[1];
  const float* pos_emb = (const float*)d_in[2];
  const float* pos_q_w = (const float*)d_in[3];
  const float* pos_q_b = (const float*)d_in[4];
  const float* pos_k_w = (const float*)d_in[5];
  const float* pos_k_b = (const float*)d_in[6];
  const float* ln_g    = (const float*)d_in[7];
  const float* ln_b    = (const float*)d_in[8];
  const float* table   = (const float*)d_in[9];
  const int*   rp      = (const int*)  d_in[10];
  const float* Wq      = (const float*)d_in[11];
  const float* bq      = (const float*)d_in[12];
  const float* Wk      = (const float*)d_in[13];
  const float* bk      = (const float*)d_in[14];
  const float* Wv      = (const float*)d_in[15];
  const float* bv      = (const float*)d_in[16];
  const float* Wo      = (const float*)d_in[17];
  const float* bo      = (const float*)d_in[18];
  float* out = (float*)d_out;

  char* p = (char*)d_ws;
  auto carve = [&](size_t bytes) -> char* {
    char* r = p;
    p += (bytes + 255) & ~(size_t)255;
    return r;
  };
  const size_t DD = (size_t)Dm * Dm;
  _Float16* xh    = (_Float16*)carve((size_t)NTOK * Dm * 2);
  _Float16* qh    = (_Float16*)carve((size_t)NTOK * Dm * 2);
  _Float16* kh    = (_Float16*)carve((size_t)NTOK * Dm * 2);
  _Float16* vh    = (_Float16*)carve((size_t)NTOK * Dm * 2);
  _Float16* oh    = (_Float16*)carve((size_t)NTOK * Dm * 2);
  float*    xf    = (float*)   carve((size_t)NTOK * Dm * 4);
  _Float16* wbuf  = (_Float16*)carve(4 * DD * 2);   // wq|wk|wv|wo (per layer)
  _Float16* posw  = (_Float16*)carve((size_t)Sm * Dm * 2);
  _Float16* pqw_h = (_Float16*)carve(DD * 2);
  _Float16* pkw_h = (_Float16*)carve(DD * 2);
  _Float16* pqh   = (_Float16*)carve((size_t)Sm * Dm * 2);
  _Float16* pkh   = (_Float16*)carve((size_t)Sm * Dm * 2);
  float*    biasb = (float*)   carve((size_t)Hm * Sm * Sm * 4);

  const int cvtBlocks = (int)((DD + 255) / 256);

  // --- positional bias path (one-time) ---
  posln_kernel<<<Sm, 256, 0, stream>>>(pos_emb, ln_g, ln_b, posw);
  cvt_f32_f16<<<cvtBlocks, 256, 0, stream>>>(pos_q_w, pqw_h, (int)DD);
  cvt_f32_f16<<<cvtBlocks, 256, 0, stream>>>(pos_k_w, pkw_h, (int)DD);
  gemm_xwt<false><<<dim3(Dm / 128, Sm / 128), 256, 0, stream>>>(
      posw, pqw_h, pos_q_b, ATT_SCALE, nullptr, pqh);
  gemm_xwt<false><<<dim3(Dm / 128, Sm / 128), 256, 0, stream>>>(
      posw, pkw_h, pos_k_b, 1.0f, nullptr, pkh);
  bias_abs_rel<<<dim3(Sm / 16, Sm / 16, Hm), 32, 0, stream>>>(
      pqh, pkh, table, rp, biasb);

  // --- token embedding gather ---
  gather_embed<<<NTOK, 256, 0, stream>>>(tokens, emb, xh);

  // --- layers ---
  const dim3 gGemm(Dm / 128, NTOK / 128);
  const dim3 gAttn(Sm / 64, Hm, Bm);
  for (int l = 0; l < Lm; ++l) {
    cvt_f32_f16<<<cvtBlocks, 256, 0, stream>>>(Wq + l * DD, wbuf + 0 * DD, (int)DD);
    cvt_f32_f16<<<cvtBlocks, 256, 0, stream>>>(Wk + l * DD, wbuf + 1 * DD, (int)DD);
    cvt_f32_f16<<<cvtBlocks, 256, 0, stream>>>(Wv + l * DD, wbuf + 2 * DD, (int)DD);
    cvt_f32_f16<<<cvtBlocks, 256, 0, stream>>>(Wo + l * DD, wbuf + 3 * DD, (int)DD);

    gemm_xwt<false><<<gGemm, 256, 0, stream>>>(xh, wbuf + 0 * DD, bq + l * Dm,
                                               ATT_SCALE, nullptr, qh);
    gemm_xwt<false><<<gGemm, 256, 0, stream>>>(xh, wbuf + 1 * DD, bk + l * Dm,
                                               1.0f, nullptr, kh);
    gemm_xwt<false><<<gGemm, 256, 0, stream>>>(xh, wbuf + 2 * DD, bv + l * Dm,
                                               1.0f, nullptr, vh);

    attn_fused<<<gAttn, 128, 0, stream>>>(qh, kh, vh, biasb, oh);

    if (l == Lm - 1) {
      gemm_xwt<true><<<gGemm, 256, 0, stream>>>(oh, wbuf + 3 * DD, bo + l * Dm,
                                                1.0f, xf, nullptr);
    } else {
      gemm_xwt<false><<<gGemm, 256, 0, stream>>>(oh, wbuf + 3 * DD, bo + l * Dm,
                                                 1.0f, nullptr, xh);
    }
  }

  copy_row0<<<Bm, 256, 0, stream>>>(xf, out);
}
